// DeformConv2d_22162031247768
// MI455X (gfx1250) — matrix operand order repaired and checked by reading, not compiled, not run on hardware
//
#include <hip/hip_runtime.h>

// ---------------------------------------------------------------------------
// DCNv2 fused im2col + WMMA GEMM for gfx1250 (MI455X), round 4.
//  - Tile = 256 Cout x 64 positions: every deformable sample is computed
//    exactly once (round-3 shape duplicated all gather/FMA work twice).
//  - K permuted tap-major (k' = tap*256 + c): bilinear metadata is
//    loop-invariant per thread across each tap's 8 K-tiles.
//  - weight pre-converted f16 + permuted into d_ws; A tiles DMA'd to LDS via
//    global_load_async_to_lds_b128 (ASYNCcnt path).
//  - B tile written as one ds_store_b128 per thread (contiguous k run).
//  - 8 back-to-back v_wmma_f32_16x16x32_f16 per wave per K-tile.
// ---------------------------------------------------------------------------

typedef __attribute__((ext_vector_type(16))) _Float16 v16h;
typedef __attribute__((ext_vector_type(8)))  _Float16 v8h;
typedef __attribute__((ext_vector_type(8)))  float    v8f;
typedef __attribute__((ext_vector_type(4)))  int      v4i;

#define NB     8
#define CIN    256
#define HH     64
#define WW     64
#define COUT   256
#define TAPS   9
#define KDIM   (CIN * TAPS)    // 2304
#define LL     (HH * WW)       // 4096
#define MTILE  256             // all Cout in one workgroup
#define LTILE  64              // position tile
#define KTILE  32
#define LDPAD  40              // LDS row pitch in halves: 80B, 16B-aligned

#if __has_builtin(__builtin_amdgcn_global_load_async_to_lds_b128)
#define HAVE_ASYNC_LDS 1
typedef __attribute__((address_space(1))) v4i GV4;   // global int4
typedef __attribute__((address_space(3))) v4i LV4;   // LDS int4
#else
#define HAVE_ASYNC_LDS 0
#endif

__device__ __forceinline__ void wait_async0() {
#if __has_builtin(__builtin_amdgcn_s_wait_asynccnt)
    __builtin_amdgcn_s_wait_asynccnt(0);
#else
    asm volatile("s_wait_asynccnt 0" ::: "memory");
#endif
}

// --- pre-pass: weight (Cout, Cin*9) f32 -> f16, permuted to k' = tap*256+c --
__global__ __launch_bounds__(256)
void wperm_kernel(const float* __restrict__ w, _Float16* __restrict__ wp) {
    const int idx = blockIdx.x * 256 + threadIdx.x;    // coalesced read
    if (idx < COUT * KDIM) {
        const int co  = idx / KDIM;
        const int k   = idx - co * KDIM;
        const int c   = k / TAPS;
        const int tap = k - c * TAPS;
        wp[(size_t)co * KDIM + tap * CIN + c] = (_Float16)w[idx];
    }
}

template <bool W16>
__global__ __launch_bounds__(256)
void dcnv2_wmma_kernel(const float* __restrict__ xin,
                       const float* __restrict__ offset,
                       const float* __restrict__ mask,
                       const float* __restrict__ weight,   // f32, !W16 path
                       const _Float16* __restrict__ wp,    // f16 permuted
                       const float* __restrict__ bias,
                       float* __restrict__ out) {
    __shared__ _Float16 sA[MTILE][LDPAD];          // 20.0 KB
    __shared__ _Float16 sB[LTILE][LDPAD];          //  5.0 KB
    __shared__ int      sIdx[TAPS][LTILE][4];      //  9.0 KB
    __shared__ float    sWgt[TAPS][LTILE][4];      //  9.0 KB

    const int tid    = threadIdx.x;
    const int wave   = tid >> 5;
    const int lane   = tid & 31;
    const int l_base = blockIdx.x * LTILE;
    const int n      = blockIdx.y;

    // ---- bilinear metadata: 9 taps x 64 positions, mask+validity folded ----
    for (int t = tid; t < TAPS * LTILE; t += 256) {
        const int tap = t >> 6;
        const int i   = t & 63;
        const int pos = l_base + i;
        const int oy  = pos >> 6;
        const int ox  = pos & 63;
        const int kh  = tap / 3;
        const int kw  = tap - kh * 3;
        const float dy = offset[(((size_t)n * 18 + tap * 2    ) << 12) + pos];
        const float dx = offset[(((size_t)n * 18 + tap * 2 + 1) << 12) + pos];
        const float mm = mask  [(((size_t)n *  9 + tap        ) << 12) + pos];
        const float ys = (float)(oy + kh - 1) + dy;    // STRIDE=PAD=DIL=1
        const float xs = (float)(ox + kw - 1) + dx;
        const float y0 = floorf(ys);
        const float x0 = floorf(xs);
        const float ly = ys - y0, lx = xs - x0;
        const float wts[4] = {(1.f - ly) * (1.f - lx), (1.f - ly) * lx,
                              ly * (1.f - lx),          ly * lx};
        const int iy0 = (int)y0, ix0 = (int)x0;
#pragma unroll
        for (int j = 0; j < 4; ++j) {
            const int yi = iy0 + (j >> 1);
            const int xi = ix0 + (j & 1);
            const bool valid = (yi >= 0) & (yi < HH) & (xi >= 0) & (xi < WW);
            const int  yc = min(max(yi, 0), HH - 1);
            const int  xc = min(max(xi, 0), WW - 1);
            sIdx[tap][i][j] = yc * WW + xc;
            sWgt[tap][i][j] = valid ? wts[j] * mm : 0.f;
        }
    }
    __syncthreads();

    v8f acc[8];                                   // [mt*4 + j]
#pragma unroll
    for (int j = 0; j < 8; ++j) acc[j] = (v8f){};

    const int lrow  = lane & 15;
    const int khalf = (lane < 16) ? 0 : 8;         // A K-split (ISA layout)
    const int kb    = (lane < 16) ? 0 : 8;         // B K-split (ISA layout)
    const int iFix  = tid & 63;                    // this thread's B column
    const int kRow  = tid >> 6;                    // 0..3: 8-wide k run

    for (int tap = 0; tap < TAPS; ++tap) {
        // metadata is loop-invariant for this thread across the whole tap
        const int   id0 = sIdx[tap][iFix][0], id1 = sIdx[tap][iFix][1];
        const int   id2 = sIdx[tap][iFix][2], id3 = sIdx[tap][iFix][3];
        const float w0  = sWgt[tap][iFix][0], w1  = sWgt[tap][iFix][1];
        const float w2  = sWgt[tap][iFix][2], w3  = sWgt[tap][iFix][3];

        for (int c0 = 0; c0 < CIN; c0 += KTILE) {
            const int k0 = tap * CIN + c0;         // permuted K base
            __syncthreads();                       // tiles free to overwrite

            // ---------- stage A: 256x32 f16 weights (16 KB) ----------
            if (W16) {
#if HAVE_ASYNC_LDS
#pragma unroll
                for (int q = 0; q < 4; ++q) {      // 1024 x 16B chunks
                    const int chunk = tid + q * 256;
                    const int m = chunk >> 2, part = chunk & 3;
                    const _Float16* g =
                        wp + (size_t)m * KDIM + k0 + part * 8;
                    __builtin_amdgcn_global_load_async_to_lds_b128(
                        (GV4*)g, (LV4*)&sA[m][part * 8], 0, 0);
                }
#else
#pragma unroll
                for (int q = 0; q < 4; ++q) {
                    const int chunk = tid + q * 256;
                    const int m = chunk >> 2, part = chunk & 3;
                    *(v8h*)&sA[m][part * 8] = *(const v8h*)(
                        wp + (size_t)m * KDIM + k0 + part * 8);
                }
#endif
            } else {
                for (int e = tid; e < MTILE * KTILE; e += 256) {
                    const int m = e >> 5, kloc = e & 31;
                    sA[m][kloc] = (_Float16)weight[
                        (size_t)m * KDIM + (size_t)(c0 + kloc) * TAPS + tap];
                }
            }

            // ---- stage B: 8 contiguous k per thread -> one ds_store_b128 ----
            {
                const float* plane =
                    xin + (((size_t)n * CIN + c0 + kRow * 8) << 12);
                v8h bvals;
#pragma unroll
                for (int s = 0; s < 8; ++s) {
                    const float v = w0 * plane[id0] + w1 * plane[id1] +
                                    w2 * plane[id2] + w3 * plane[id3];
                    bvals[s] = (_Float16)v;
                    plane += LL;
                }
                *(v8h*)&sB[iFix][kRow * 8] = bvals;
            }

#if HAVE_ASYNC_LDS
            if (W16) wait_async0();
#endif
            __syncthreads();

            // ---------- fragments: batch DS loads, then 8 WMMAs ----------
            v16h afrag[2];
#pragma unroll
            for (int mt = 0; mt < 2; ++mt) {
                const int mr = mt * 128 + (wave << 4) + lrow;
                const v8h alo = *(const v8h*)&sA[mr][khalf];
                const v8h ahi = *(const v8h*)&sA[mr][khalf + 16];
#pragma unroll
                for (int q = 0; q < 8; ++q) {
                    afrag[mt][q] = alo[q]; afrag[mt][q + 8] = ahi[q];
                }
            }
            v16h bf[4];
#pragma unroll
            for (int j = 0; j < 4; ++j) {
                const int prow = (j << 4) + lrow;
                const v8h blo = *(const v8h*)&sB[prow][kb];
                const v8h bhi = *(const v8h*)&sB[prow][kb + 16];
#pragma unroll
                for (int q = 0; q < 8; ++q) {
                    bf[j][q] = blo[q]; bf[j][q + 8] = bhi[q];
                }
            }
#pragma unroll
            for (int mt = 0; mt < 2; ++mt)
#pragma unroll
                for (int j = 0; j < 4; ++j)
                    acc[mt * 4 + j] = __builtin_amdgcn_wmma_f32_16x16x32_f16(
                        false, afrag[mt], false, bf[j], (short)0,
                        acc[mt * 4 + j], false, false);
        }
    }

    // ---- epilogue: C layout lanes 0-15 -> M=r, lanes 16-31 -> M=r+8 ----
    const int mhalf = (lane < 16) ? 0 : 8;
    const int ncol  = lane & 15;
#pragma unroll
    for (int mt = 0; mt < 2; ++mt) {
#pragma unroll
        for (int j = 0; j < 4; ++j) {
            const int pos = l_base + (j << 4) + ncol;
#pragma unroll
            for (int r = 0; r < 8; ++r) {
                const int co = mt * 128 + (wave << 4) + r + mhalf;
                out[(((size_t)n * COUT + co) << 12) + pos] =
                    acc[mt * 4 + j][r] + bias[co];
            }
        }
    }
}

extern "C" void kernel_launch(void* const* d_in, const int* in_sizes, int n_in,
                              void* d_out, int out_size, void* d_ws, size_t ws_size,
                              hipStream_t stream) {
    const float* xin    = (const float*)d_in[0];
    const float* offset = (const float*)d_in[1];
    const float* mask   = (const float*)d_in[2];
    const float* weight = (const float*)d_in[3];
    const float* bias   = (const float*)d_in[4];
    float* out = (float*)d_out;

    dim3 grid(LL / LTILE, NB);                 // (64, 8) = 512 workgroups
    dim3 block(256);

    const size_t wpBytes = (size_t)COUT * KDIM * sizeof(_Float16); // 1.18 MB
    if (ws_size >= wpBytes) {
        _Float16* wp = (_Float16*)d_ws;
        wperm_kernel<<<(COUT * KDIM + 255) / 256, 256, 0, stream>>>(weight, wp);
        dcnv2_wmma_kernel<true><<<grid, block, 0, stream>>>(
            xin, offset, mask, weight, wp, bias, out);
    } else {
        dcnv2_wmma_kernel<false><<<grid, block, 0, stream>>>(
            xin, offset, mask, weight, nullptr, bias, out);
    }
}
